// PI0Policy_14190571946468
// MI455X (gfx1250) — matrix operand order repaired and checked by reading, not compiled, not run on hardware
//
#include <hip/hip_runtime.h>
#include <cstdint>
#include <cstddef>

#ifndef __has_builtin
#define __has_builtin(x) 0
#endif

// ---------------------------------------------------------------------------
// Problem constants (from reference)
// ---------------------------------------------------------------------------
static const int Bc  = 16;
static const int L1c = 816;
static const int L2c = 51;
static const int Nc  = L1c + L2c;   // 867
static const int Hc  = 8;
static const int DHc = 256;
static const int D1c = 2048;
static const int D2c = 1024;
static const int HDc = Hc * DHc;    // 2048

typedef __attribute__((ext_vector_type(16))) __bf16 v16bf;
typedef __attribute__((ext_vector_type(8)))  float  v8f;

// ---------------------------------------------------------------------------
// Async global->LDS copy (CDNA5 GLOBAL_LOAD_ASYNC_TO_LDS_B128, ASYNCcnt)
// Builtin signature (from clang diagnostic): (v4i32 AS1*, v4i32 AS3*, Ii, Ii)
// ---------------------------------------------------------------------------
typedef __attribute__((__vector_size__(4 * sizeof(int)))) int v4i32;
typedef __attribute__((address_space(1))) v4i32 gv4i32;
typedef __attribute__((address_space(3))) v4i32 lv4i32;

#if __has_builtin(__builtin_amdgcn_global_load_async_to_lds_b128)
#define HAVE_ASYNC_LDS 1
#define ASYNC_CP16(ldst, gsrc)                                              \
    __builtin_amdgcn_global_load_async_to_lds_b128(                         \
        (gv4i32*)(void*)(uintptr_t)(gsrc),                                  \
        (lv4i32*)(void*)(ldst), 0, 0)
#if __has_builtin(__builtin_amdgcn_s_wait_asynccnt)
#define ASYNC_WAIT_N(n) __builtin_amdgcn_s_wait_asynccnt(n)
#else
#define ASYNC_WAIT_N(n) asm volatile("s_wait_asynccnt %0" ::"i"(n) : "memory")
#endif
#else
#define HAVE_ASYNC_LDS 0
#define ASYNC_CP16(ldst, gsrc)                                              \
    (*reinterpret_cast<uint4*>(ldst) = *reinterpret_cast<const uint4*>(gsrc))
#define ASYNC_WAIT_N(n) ((void)0)
#endif

// ---------------------------------------------------------------------------
// WMMA fragment loaders (wave32; A/B use contiguous k = 16*hf + s, not the ISA table order).
// A 16x32 bf16 per lane: VGPR0..3 cover k = kb + hf*16 + [0..7]  (16 contiguous B)
//                        VGPR4..7 cover k = kb + hf*16 + 8 + [0..7]
//   -> two ds_load_b128 per fragment.
// B 32x16 bf16 per lane (n = lane&15): pair kp = hf*8 + v; from n-major
//   pair-interleaved storage Bq[n][kp] -> 8 contiguous u32 -> two ds_load_b128.
// C/D 16x16 f32: lane n = lane&15; VGPR v row M = v + 8*(lane>>4)
// ---------------------------------------------------------------------------
__device__ __forceinline__ v16bf load_a_frag(const __bf16* S, int row_base, int ld,
                                             int k_base, int lane) {
    const int m  = row_base + (lane & 15);
    const int hf = lane >> 4;
    const unsigned* row =
        reinterpret_cast<const unsigned*>(S + m * ld) + (k_base >> 1) + hf * 8;
    union { v16bf v; uint4 q[2]; } a;
    a.q[0] = *reinterpret_cast<const uint4*>(row);
    a.q[1] = *reinterpret_cast<const uint4*>(row + 4);
    return a.v;
}

// B fragment from n-major pair-interleaved storage (pitch ldq in u32)
__device__ __forceinline__ v16bf load_b_fragq(const unsigned* Bq, int ldq,
                                              int n_base, int kp_base, int lane) {
    const int n  = n_base + (lane & 15);
    const int hf = lane >> 4;
    const unsigned* row = Bq + n * ldq + kp_base + hf * 8;
    union { v16bf v; uint4 q[2]; } b;
    b.q[0] = *reinterpret_cast<const uint4*>(row);
    b.q[1] = *reinterpret_cast<const uint4*>(row + 4);
    return b.v;
}

// B fragment from transposed row-major storage: B[k][n] = S[n*ld + k]
// (16 contiguous k values per lane -> two ds_load_b128)
__device__ __forceinline__ v16bf load_bT_frag(const __bf16* S, int n_base, int ld,
                                              int k_base, int lane) {
    const int n  = n_base + (lane & 15);
    const int hf = lane >> 4;
    const unsigned* row =
        reinterpret_cast<const unsigned*>(S + n * ld) + (k_base >> 1) + hf * 8;
    union { v16bf v; uint4 q[2]; } b;
    b.q[0] = *reinterpret_cast<const uint4*>(row);
    b.q[1] = *reinterpret_cast<const uint4*>(row + 4);
    return b.v;
}

// key-pair interleaved V: P[kp][n] packs keys (2kp, 2kp+1) at column n
__device__ __forceinline__ v16bf load_b_frag_pi(const unsigned* P, int kp_base,
                                                int ldp, int n_base, int lane) {
    const int n  = n_base + (lane & 15);
    const int hf = lane >> 4;
    union { v16bf v; unsigned u[8]; } b;
#pragma unroll
    for (int v = 0; v < 8; ++v)
        b.u[v] = P[(kp_base + hf * 8 + v) * ldp + n];
    return b.v;
}

// ---------------------------------------------------------------------------
// Weight cast f32 -> bf16, n-major pair-interleaved: out[n*(K/2) + kp] packs
// rows 2kp (lo16), 2kp+1 (hi16) of column n. B staging becomes a contiguous
// copy AND B fragments become two b128 LDS loads.
// ---------------------------------------------------------------------------
__global__ void cast_interleave_nmajor(const float* __restrict__ in,
                                       unsigned* __restrict__ out, int K, int N) {
    const size_t total = (size_t)N * (K / 2);
    size_t i = (size_t)blockIdx.x * blockDim.x + threadIdx.x;
    if (i >= total) return;
    const int kp = (int)(i % (K / 2));
    const int n  = (int)(i / (K / 2));
    union { __bf16 h[2]; unsigned u; } pk;
    pk.h[0] = (__bf16)in[(size_t)(2 * kp) * N + n];
    pk.h[1] = (__bf16)in[(size_t)(2 * kp + 1) * N + n];
    out[i] = pk.u;
}

// ---------------------------------------------------------------------------
// RMS norm + cast: one block per token
// ---------------------------------------------------------------------------
__global__ void __launch_bounds__(256) rmsnorm_cast(const float* __restrict__ x,
                                                    const float* __restrict__ w,
                                                    __bf16* __restrict__ out, int D) {
    const size_t t = blockIdx.x;
    __shared__ float red[256];
    float ss = 0.f;
    for (int i = threadIdx.x; i < D; i += 256) {
        float v = x[t * D + i];
        ss += v * v;
    }
    red[threadIdx.x] = ss;
    __syncthreads();
    for (int s = 128; s > 0; s >>= 1) {
        if ((int)threadIdx.x < s) red[threadIdx.x] += red[threadIdx.x + s];
        __syncthreads();
    }
    const float scale = rsqrtf(red[0] / (float)D + 1e-6f);
    for (int i = threadIdx.x; i < D; i += 256)
        out[t * D + i] = (__bf16)(x[t * D + i] * scale * (1.0f + w[i]));
}

// ---------------------------------------------------------------------------
// bf16 WMMA GEMM, double-buffered async pipeline:
//   C[M,N](+row remap) = A[M,K](bf16) * Wq[N, K/2](u32 pairs)
// Block tile 256x128, BK=32. 8 waves (4 row x 2 col), wave tile 64x64
// (4x4 WMMA tiles, 128 f32 accum VGPRs/wave). LDS 52 KB.
// Per-wave async transfers/tile: 4 (A) + 2 (B) = 6 -> s_wait_asynccnt 6
// overlaps tile t+1 fetch with tile t compute.
// ---------------------------------------------------------------------------
#define BM 256
#define BN 128
#define BK 32
#define BKH 16

template <typename OutT>
__global__ void __launch_bounds__(256) gemm_bf16(
    const __bf16* __restrict__ A, const unsigned* __restrict__ Wq,
    OutT* __restrict__ C, int M, int N, int K,
    int inL, int inStride, int inOff,
    int outL, int outStride, int outOff) {
    __shared__ __bf16   As[2][BM][BK];        // 32 KB
    __shared__ unsigned Bq[2][BN][BKH + 4];   // 20 KB, pitch 80 B

    const int lane = threadIdx.x & 31;
    const int wave = threadIdx.x >> 5;     // 0..7
    const int wm = wave & 3;               // row block of 64
    const int wn = wave >> 2;              // col block of 64
    const int block_m = blockIdx.x * BM;
    const int block_n = blockIdx.y * BN;

    auto stage = [&](int buf, int k0) {
        // A tile: 256x32 bf16, contiguous 16B per lane (4 instr/wave)
        for (int i = threadIdx.x; i < (BM * BK) / 8; i += 256) {
            const int r = (i * 8) / BK;
            const int c = (i * 8) % BK;
            int gm = block_m + r;
            gm = gm < M ? gm : M - 1;  // clamp: rows >= M never stored
            const int grow = (gm / inL) * inStride + inOff + (gm % inL);
            ASYNC_CP16(&As[buf][r][c], A + (size_t)grow * K + k0 + c);
        }
        // B tile: 128 rows x 16 u32 pairs, contiguous (2 instr/wave)
        for (int i = threadIdx.x; i < (BN * BKH) / 4; i += 256) {
            const int n  = (i * 4) / BKH;
            const int cp = (i * 4) % BKH;
            ASYNC_CP16(&Bq[buf][n][cp],
                       Wq + (size_t)(block_n + n) * (K / 2) + (k0 >> 1) + cp);
        }
    };

    v8f acc[4][4] = {};
    stage(0, 0);
    const int ntiles = K / BK;

    for (int t = 0; t < ntiles; ++t) {
        const int cur = t & 1;
        __syncthreads();                      // all waves done with buffer cur^1
        if (t + 1 < ntiles) {
            stage(cur ^ 1, (t + 1) * BK);     // prefetch next tile
            ASYNC_WAIT_N(6);                  // tile t (older 6/wave) complete
        } else {
            ASYNC_WAIT_N(0);
        }
        __syncthreads();                      // tile t visible to all waves

        v16bf af[4], bfr[4];
#pragma unroll
        for (int i = 0; i < 4; ++i)
            af[i] = load_a_frag(&As[cur][0][0], wm * 64 + i * 16, BK, 0, lane);
#pragma unroll
        for (int j = 0; j < 4; ++j)
            bfr[j] = load_b_fragq(&Bq[cur][0][0], BKH + 4, wn * 64 + j * 16, 0,
                                  lane);
#pragma unroll
        for (int i = 0; i < 4; ++i)
#pragma unroll
            for (int j = 0; j < 4; ++j)
                acc[i][j] = __builtin_amdgcn_wmma_f32_16x16x32_bf16(
                    false, af[i], false, bfr[j], (short)0, acc[i][j],
                    false, false);
    }

    // store (C/D layout: n = lane&15, row M = v + 8*(lane>>4))
    const int hf = lane >> 4;
    const int nl = lane & 15;
#pragma unroll
    for (int i = 0; i < 4; ++i)
#pragma unroll
        for (int j = 0; j < 4; ++j) {
            const int n = block_n + wn * 64 + j * 16 + nl;
            if (n >= N) continue;
#pragma unroll
            for (int v = 0; v < 8; ++v) {
                const int m = block_m + wm * 64 + i * 16 + v + 8 * hf;
                if (m >= M) continue;
                const int orow = (m / outL) * outStride + outOff + (m % outL);
                C[(size_t)orow * N + n] = (OutT)acc[i][j][v];
            }
        }
}

// ---------------------------------------------------------------------------
// RoPE in place on bf16 buffer [B*N, H*DH]; pair (i, i+DH/2) per thread.
// ---------------------------------------------------------------------------
__global__ void __launch_bounds__(128) rope_kernel(__bf16* __restrict__ x,
                                                   const int* __restrict__ pos_ids) {
    const int row = blockIdx.x;          // b*Nc + l
    const int h   = blockIdx.y;
    const int i   = threadIdx.x;         // 0..127
    const float pos = (float)pos_ids[row];
    const float fe  = (2.0f / (float)DHc) * (float)i;
    const float ts  = powf(10000.0f, fe);
    const float rad = pos / ts;
    float s, c;
    __sincosf(rad, &s, &c);
    const size_t base = (size_t)row * HDc + (size_t)h * DHc;
    const float xa = (float)x[base + i];
    const float xb = (float)x[base + i + DHc / 2];
    x[base + i]           = (__bf16)(xa * c - xb * s);
    x[base + i + DHc / 2] = (__bf16)(xb * c + xa * s);
}

// ---------------------------------------------------------------------------
// Mask cumsum per batch row
// ---------------------------------------------------------------------------
__global__ void cumsum_kernel(const int* __restrict__ att, int* __restrict__ cums) {
    const int b = blockIdx.x * blockDim.x + threadIdx.x;
    if (b >= Bc) return;
    int c = 0;
    for (int i = 0; i < Nc; ++i) {
        c += att[b * Nc + i];
        cums[b * Nc + i] = c;
    }
}

// ---------------------------------------------------------------------------
// Flash attention: one wave per (16-row q tile, head, batch).
// QK^T and PV via v_wmma_f32_16x16x32_bf16; Q + double-buffered K async.
// ---------------------------------------------------------------------------
__global__ void __launch_bounds__(32) flash_attn(
    const __bf16* __restrict__ Q, const __bf16* __restrict__ Kb,
    const __bf16* __restrict__ Vb, __bf16* __restrict__ O,
    const int* __restrict__ cums, const unsigned char* __restrict__ pad) {
    const int qt   = blockIdx.x;
    const int h    = blockIdx.y;
    const int b    = blockIdx.z;
    const int lane = threadIdx.x;
    const int q0   = qt * 16;
    const int hf   = lane >> 4;
    const int nl   = lane & 15;

    __shared__ __bf16   Qs[16][DHc + 8];      // pitch 528 B
    __shared__ __bf16   Ks[2][32][DHc + 8];   // double-buffered K tiles
    __shared__ unsigned Vp[16][DHc + 4];      // key-pair interleaved V
    __shared__ __bf16   Ps[16][32];
    __shared__ int      qc[16];

    auto stageK = [&](int buf, int kt) {      // 32 async instr/wave
        for (int idx = lane; idx < (32 * DHc) / 8; idx += 32) {
            const int r = (idx * 8) / DHc;
            const int c = (idx * 8) % DHc;
            int krow = kt + r;
            krow = krow < Nc ? krow : Nc - 1;
            ASYNC_CP16(&Ks[buf][r][c],
                       Kb + (size_t)(b * Nc + krow) * HDc + (size_t)h * DHc + c);
        }
    };

    // Q tile (rows clamped; clamped rows never stored), then prefetch K tile 0
    for (int idx = lane; idx < (16 * DHc) / 8; idx += 32) {
        const int r = (idx * 8) / DHc;
        const int c = (idx * 8) % DHc;
        int qrow = q0 + r;
        qrow = qrow < Nc ? qrow : Nc - 1;
        ASYNC_CP16(&Qs[r][c],
                   Q + (size_t)(b * Nc + qrow) * HDc + (size_t)h * DHc + c);
    }
    stageK(0, 0);
    if (lane < 16)
        qc[lane] = (q0 + lane < Nc) ? cums[b * Nc + q0 + lane] : 0x7fffffff;

    v8f Oacc[16] = {};
    float mrow[8], lrow[8];
#pragma unroll
    for (int v = 0; v < 8; ++v) {
        mrow[v] = -1e30f;
        lrow[v] = 0.f;
    }

    int ti = 0;
    for (int kt = 0; kt < Nc; kt += 32, ++ti) {
        const int cur = ti & 1;
        if (kt + 32 < Nc) {
            stageK(cur ^ 1, kt + 32);   // prefetch next K tile
            ASYNC_WAIT_N(32);           // current tile (older) complete
        } else {
            ASYNC_WAIT_N(0);
        }
        // stage V tile pair-interleaved over keys (register pack, b128 stores)
        for (int idx = lane; idx < (16 * DHc) / 8; idx += 32) {
            const int rp = (idx * 8) / DHc;  // key pair
            const int c  = (idx * 8) % DHc;
            int r0 = kt + 2 * rp, r1 = r0 + 1;
            r0 = r0 < Nc ? r0 : Nc - 1;
            r1 = r1 < Nc ? r1 : Nc - 1;
            const uint4 va = *reinterpret_cast<const uint4*>(
                Vb + (size_t)(b * Nc + r0) * HDc + (size_t)h * DHc + c);
            const uint4 vb = *reinterpret_cast<const uint4*>(
                Vb + (size_t)(b * Nc + r1) * HDc + (size_t)h * DHc + c);
            const unsigned ua[4] = {va.x, va.y, va.z, va.w};
            const unsigned ub[4] = {vb.x, vb.y, vb.z, vb.w};
            unsigned res[8];
#pragma unroll
            for (int j = 0; j < 4; ++j) {
                res[2 * j]     = (ua[j] & 0xFFFFu) | (ub[j] << 16);
                res[2 * j + 1] = (ua[j] >> 16) | (ub[j] & 0xFFFF0000u);
            }
            uint4 o0 = {res[0], res[1], res[2], res[3]};
            uint4 o1 = {res[4], res[5], res[6], res[7]};
            *reinterpret_cast<uint4*>(&Vp[rp][c])     = o0;
            *reinterpret_cast<uint4*>(&Vp[rp][c + 4]) = o1;
        }
        __syncthreads();

        // S = Q * K^T  (two 16x16 C tiles over the 32 keys)
        v8f S0 = {}, S1 = {};
#pragma unroll
        for (int dk = 0; dk < DHc; dk += 32) {
            v16bf aq = load_a_frag(&Qs[0][0], 0, DHc + 8, dk, lane);
            v16bf b0 = load_bT_frag(&Ks[cur][0][0], 0,  DHc + 8, dk, lane);
            v16bf b1 = load_bT_frag(&Ks[cur][0][0], 16, DHc + 8, dk, lane);
            S0 = __builtin_amdgcn_wmma_f32_16x16x32_bf16(false, aq, false, b0,
                                                         (short)0, S0, false, false);
            S1 = __builtin_amdgcn_wmma_f32_16x16x32_bf16(false, aq, false, b1,
                                                         (short)0, S1, false, false);
        }

        // mask + online softmax (row M = v + 8*hf lives in one 16-lane half)
        const int key0 = kt + nl;
        const int key1 = kt + 16 + nl;
        const bool ok0 = (key0 < Nc) && pad[b * Nc + key0];
        const bool ok1 = (key1 < Nc) && pad[b * Nc + key1];
        const int kc0 = ok0 ? cums[b * Nc + key0] : 0x7fffffff;
        const int kc1 = ok1 ? cums[b * Nc + key1] : 0x7fffffff;
        const float scale = 0.0625f;  // DH^-0.5
#pragma unroll
        for (int v = 0; v < 8; ++v) {
            const int M = v + 8 * hf;
            const int qcv = qc[M];
            float s0 = (ok0 && kc0 <= qcv) ? S0[v] * scale : -1e30f;
            float s1 = (ok1 && kc1 <= qcv) ? S1[v] * scale : -1e30f;
            float mx = fmaxf(s0, s1);
#pragma unroll
            for (int off = 1; off < 16; off <<= 1)
                mx = fmaxf(mx, __shfl_xor(mx, off, 32));
            const float mnew = fmaxf(mrow[v], mx);
            const float corr = __expf(mrow[v] - mnew);
            const float p0 = __expf(s0 - mnew);
            const float p1 = __expf(s1 - mnew);
            float rs = p0 + p1;
#pragma unroll
            for (int off = 1; off < 16; off <<= 1) rs += __shfl_xor(rs, off, 32);
            lrow[v] = lrow[v] * corr + rs;
            mrow[v] = mnew;
            Ps[M][nl]      = (__bf16)p0;
            Ps[M][16 + nl] = (__bf16)p1;
#pragma unroll
            for (int j = 0; j < 16; ++j) Oacc[j][v] *= corr;
        }
        __syncthreads();

        // O += P * V  (A = P 16x32 via two b128 loads, B = pair-interleaved V)
        v16bf ap = load_a_frag(&Ps[0][0], 0, 32, 0, lane);
#pragma unroll
        for (int j = 0; j < 16; ++j) {
            v16bf bv = load_b_frag_pi(&Vp[0][0], 0, DHc + 4, j * 16, lane);
            Oacc[j] = __builtin_amdgcn_wmma_f32_16x16x32_bf16(
                false, ap, false, bv, (short)0, Oacc[j], false, false);
        }
        __syncthreads();
    }

    // normalize + store bf16 [B*Nc, H*DH]
#pragma unroll
    for (int j = 0; j < 16; ++j) {
#pragma unroll
        for (int v = 0; v < 8; ++v) {
            const int qrow = q0 + v + 8 * hf;
            if (qrow >= Nc) continue;
            const float val = Oacc[j][v] / lrow[v];
            O[(size_t)(b * Nc + qrow) * HDc + (size_t)h * DHc + j * 16 + nl] =
                (__bf16)val;
        }
    }
}

// ---------------------------------------------------------------------------
// Host orchestration
// ---------------------------------------------------------------------------
extern "C" void kernel_launch(void* const* d_in, const int* in_sizes, int n_in,
                              void* d_out, int out_size, void* d_ws, size_t ws_size,
                              hipStream_t stream) {
    (void)in_sizes; (void)n_in; (void)out_size; (void)ws_size;

    const float* x1  = (const float*)d_in[0];
    const float* x2  = (const float*)d_in[1];
    const float* ln1 = (const float*)d_in[2];
    const float* ln2 = (const float*)d_in[3];
    const float* W[8] = {
        (const float*)d_in[4],  (const float*)d_in[5],  (const float*)d_in[6],
        (const float*)d_in[7],  (const float*)d_in[8],  (const float*)d_in[9],
        (const float*)d_in[10], (const float*)d_in[11]};
    const int* pos           = (const int*)d_in[12];
    const unsigned char* pad = (const unsigned char*)d_in[13];
    const int* att           = (const int*)d_in[14];

    // workspace bump allocator
    char* p = (char*)d_ws;
    auto alloc = [&](size_t bytes) -> char* {
        char* r = p;
        p += (bytes + 255) & ~(size_t)255;
        return r;
    };

    // weight (K, N) per input; n-major interleaved u32 buffers: N*(K/2) words
    const int wK[8] = {D1c, D1c, D1c, HDc, D2c, D2c, D2c, HDc};
    const int wN[8] = {HDc, HDc, HDc, D1c, HDc, HDc, HDc, D2c};

    __bf16* h1 = (__bf16*)alloc((size_t)Bc * L1c * D1c * 2);
    __bf16* h2 = (__bf16*)alloc((size_t)Bc * L2c * D2c * 2);
    unsigned* Wb[8];
    for (int i = 0; i < 8; ++i)
        Wb[i] = (unsigned*)alloc((size_t)wN[i] * (wK[i] / 2) * 4);
    __bf16* qbuf = (__bf16*)alloc((size_t)Bc * Nc * HDc * 2);
    __bf16* kbuf = (__bf16*)alloc((size_t)Bc * Nc * HDc * 2);
    __bf16* vbuf = (__bf16*)alloc((size_t)Bc * Nc * HDc * 2);
    __bf16* obuf = (__bf16*)alloc((size_t)Bc * Nc * HDc * 2);
    int*    cums = (int*)alloc((size_t)Bc * Nc * 4);

    // 1) cast + n-major pair-interleave weights
    for (int i = 0; i < 8; ++i) {
        const size_t n = (size_t)wN[i] * (wK[i] / 2);
        cast_interleave_nmajor<<<dim3((unsigned)((n + 255) / 256)), dim3(256), 0,
                                 stream>>>(W[i], Wb[i], wK[i], wN[i]);
    }

    // 2) RMS norm + cast activations
    rmsnorm_cast<<<dim3(Bc * L1c), dim3(256), 0, stream>>>(x1, ln1, h1, D1c);
    rmsnorm_cast<<<dim3(Bc * L2c), dim3(256), 0, stream>>>(x2, ln2, h2, D2c);

    // 3) QKV projections -> scattered into [B, Nc, H*DH]
    const int M1 = Bc * L1c, M2 = Bc * L2c;
    {
        dim3 g1((M1 + BM - 1) / BM, (HDc + BN - 1) / BN), blk(256);
        gemm_bf16<__bf16><<<g1, blk, 0, stream>>>(h1, Wb[0], qbuf, M1, HDc, D1c,
                                                  M1, 0, 0, L1c, Nc, 0);
        gemm_bf16<__bf16><<<g1, blk, 0, stream>>>(h1, Wb[1], kbuf, M1, HDc, D1c,
                                                  M1, 0, 0, L1c, Nc, 0);
        gemm_bf16<__bf16><<<g1, blk, 0, stream>>>(h1, Wb[2], vbuf, M1, HDc, D1c,
                                                  M1, 0, 0, L1c, Nc, 0);
        dim3 g2((M2 + BM - 1) / BM, (HDc + BN - 1) / BN);
        gemm_bf16<__bf16><<<g2, blk, 0, stream>>>(h2, Wb[4], qbuf, M2, HDc, D2c,
                                                  M2, 0, 0, L2c, Nc, L1c);
        gemm_bf16<__bf16><<<g2, blk, 0, stream>>>(h2, Wb[5], kbuf, M2, HDc, D2c,
                                                  M2, 0, 0, L2c, Nc, L1c);
        gemm_bf16<__bf16><<<g2, blk, 0, stream>>>(h2, Wb[6], vbuf, M2, HDc, D2c,
                                                  M2, 0, 0, L2c, Nc, L1c);
    }

    // 4) RoPE on q and k
    {
        dim3 g(Bc * Nc, Hc), blk(128);
        rope_kernel<<<g, blk, 0, stream>>>(qbuf, pos);
        rope_kernel<<<g, blk, 0, stream>>>(kbuf, pos);
    }

    // 5) mask cumsum
    cumsum_kernel<<<dim3(1), dim3(32), 0, stream>>>(att, cums);

    // 6) attention
    {
        dim3 g((Nc + 15) / 16, Hc, Bc), blk(32);
        flash_attn<<<g, blk, 0, stream>>>(qbuf, kbuf, vbuf, obuf, cums, pad);
    }

    // 7) output projections -> f32 d_out (o1 then o2, flat)
    float* out = (float*)d_out;
    float* o2p = out + (size_t)Bc * L1c * D1c;
    {
        dim3 blk(256);
        dim3 g1((M1 + BM - 1) / BM, (D1c + BN - 1) / BN);
        gemm_bf16<float><<<g1, blk, 0, stream>>>(obuf, Wb[3], out, M1, D1c, HDc,
                                                 L1c, Nc, 0, M1, 0, 0);
        dim3 g2((M2 + BM - 1) / BM, (D2c + BN - 1) / BN);
        gemm_bf16<float><<<g2, blk, 0, stream>>>(obuf, Wb[7], o2p, M2, D2c, HDc,
                                                 L2c, Nc, L1c, M2, 0, 0);
    }
}
